// OuterTransformer_60962765799601
// MI455X (gfx1250) — matrix operand order repaired and checked by reading, not compiled, not run on hardware
//
#include <hip/hip_runtime.h>

// ---------------------------------------------------------------------------
// OuterTransformer on MI455X (gfx1250, wave32, WMMA)
//   G   = Xf^T Xf            (Xf = x flattened [N,D], N=32768, D=256)
//   M   = G W + colsum(Xf) b^T
//   Out = Xf M
// Split-bf16 (hi+lo) emulated-fp32 GEMMs on v_wmma_f32_16x16x32_bf16.
// Async global->LDS staging (ASYNCcnt) for the M^T operand in kernel 3.
// ---------------------------------------------------------------------------

typedef __bf16 bf16;
typedef __bf16 v16bf __attribute__((ext_vector_type(16)));
typedef float  v8f   __attribute__((ext_vector_type(8)));

#define N_TOT  32768
#define DDIM   256
#define P_GRAM 64
#define CHUNK  (N_TOT / P_GRAM)   // 512 rows per gram workgroup
#define KSTEP  128                // K-slab rows staged in LDS per pass
#define RS     136                // padded LDS row stride in bf16 (272B = 68 dwords -> 4-bank rotate/row)

// ---- fragment loaders (ISA 7.12.2 layouts, wave32) ------------------------
// A matrix 16x32 bf16, row-major [m][k], stride RS:
//   lane: m = lane&15, h = lane>>4
//   elems 0..7  : K = 8h + 0..7      ; elems 8..15 : K = 16 + 8h + 0..7
__device__ __forceinline__ v16bf frag_a(const bf16* base, int lane) {
  int m = lane & 15, h = lane >> 4;
  const bf16* p = base + m * RS + 8 * h;
  union { v16bf v; uint4 q[2]; } u;
  u.q[0] = *(const uint4*)(p);
  u.q[1] = *(const uint4*)(p + 16);
  return u.v;
}
// B matrix 32x16 bf16, stored TRANSPOSED [n][k], stride RS:
//   lane: n = lane&15, h = lane>>4 ; elems 0..7 : K = 8h + 0..7 ; elems 8..15 : K = 16 + 8h + 0..7
__device__ __forceinline__ v16bf frag_b(const bf16* base, int lane) {
  int n = lane & 15, h = lane >> 4;
  const bf16* p = base + n * RS + 8 * h;
  union { v16bf v; uint4 q[2]; } u;
  u.q[0] = *(const uint4*)(p);
  u.q[1] = *(const uint4*)(p + 16);
  return u.v;
}

// split-fp32 product: acc += (Ah+Al)*(Bh+Bl) ~ Ah*Bh + Ah*Bl + Al*Bh
__device__ __forceinline__ v8f wmma3(v16bf ah, v16bf al, v16bf bh, v16bf bl, v8f c) {
  c = __builtin_amdgcn_wmma_f32_16x16x32_bf16(false, ah, false, bh, (short)0, c, false, false);
  c = __builtin_amdgcn_wmma_f32_16x16x32_bf16(false, ah, false, bl, (short)0, c, false, false);
  c = __builtin_amdgcn_wmma_f32_16x16x32_bf16(false, al, false, bh, (short)0, c, false, false);
  return c;
}

// ---------------------------------------------------------------------------
// Kernel 1: partial Gram  wsG[p] = Xchunk^T Xchunk  (+ partial column sums)
// grid = P_GRAM blocks x 1024 threads (32 waves). Each wave owns 8 of the
// 256 16x16 output tiles. 384 WMMAs / wave. Staging packs 4 k-adjacent bf16
// per ds_store_b64 (4x fewer DS stores than element-wise b16 stores).
// ---------------------------------------------------------------------------
__global__ __launch_bounds__(1024, 1)
void gram_kernel(const float* __restrict__ x,
                 float* __restrict__ wsG, float* __restrict__ wsS) {
  __shared__ bf16 sh[2][DDIM][RS];          // [hi/lo][d][k]  139264 B
  const int t    = threadIdx.x;
  const int lane = t & 31;
  const int w    = t >> 5;
  const int ti   = w >> 1;
  const int tj0  = (w & 1) * 8;
  const int rowBase = blockIdx.x * CHUNK;
  const int c4   = (t & 63) * 4;            // this thread's 4 columns (constant)
  const int g    = t >> 6;                  // row-group 0..15 (8 rows each)

  v8f acc[8] = {};
  float csum[4] = {0.f, 0.f, 0.f, 0.f};

  for (int s = 0; s < CHUNK / KSTEP; ++s) {
    __syncthreads();
    // stage 128x256 fp32 -> transposed hi/lo bf16 planes [d][k], packed b64
#pragma unroll
    for (int r = 0; r < 2; ++r) {
      const int kk = g * 8 + r * 4;         // 4 consecutive k-rows
      const float* src = x + (size_t)(rowBase + s * KSTEP + kk) * DDIM + c4;
      float4 f0 = *(const float4*)(src);
      float4 f1 = *(const float4*)(src + DDIM);
      float4 f2 = *(const float4*)(src + 2 * DDIM);
      float4 f3 = *(const float4*)(src + 3 * DDIM);
      float fv[4][4] = {{f0.x, f0.y, f0.z, f0.w},
                        {f1.x, f1.y, f1.z, f1.w},
                        {f2.x, f2.y, f2.z, f2.w},
                        {f3.x, f3.y, f3.z, f3.w}};
#pragma unroll
      for (int jj = 0; jj < 4; ++jj) {
        union { uint2 u; bf16 e[4]; } H, L;
#pragma unroll
        for (int rr = 0; rr < 4; ++rr) {
          float v = fv[rr][jj];
          bf16 hi = (bf16)v;
          H.e[rr] = hi;
          L.e[rr] = (bf16)(v - (float)hi);
          csum[jj] += v;
        }
        *(uint2*)&sh[0][c4 + jj][kk] = H.u;   // 8B aligned: (c4+jj)*272 + 2*kk
        *(uint2*)&sh[1][c4 + jj][kk] = L.u;
      }
    }
    __syncthreads();
#pragma unroll
    for (int ks = 0; ks < KSTEP / 32; ++ks) {
      v16bf ah = frag_a(&sh[0][ti * 16][ks * 32], lane);
      v16bf al = frag_a(&sh[1][ti * 16][ks * 32], lane);
#pragma unroll
      for (int j = 0; j < 8; ++j) {
        v16bf bh = frag_b(&sh[0][(tj0 + j) * 16][ks * 32], lane);
        v16bf bl = frag_b(&sh[1][(tj0 + j) * 16][ks * 32], lane);
        acc[j] = wmma3(ah, al, bh, bl, acc[j]);
      }
    }
  }

  // write partial Gram (C/D layout: row = q + 8h, col = lane&15)
  {
    float* gp = wsG + (size_t)blockIdx.x * DDIM * DDIM;
    int h = lane >> 4, n = lane & 15;
#pragma unroll
    for (int j = 0; j < 8; ++j) {
      int col = (tj0 + j) * 16 + n;
#pragma unroll
      for (int q = 0; q < 8; ++q)
        gp[(ti * 16 + q + 8 * h) * DDIM + col] = acc[j][q];
    }
  }

  // partial column sums (bias outer-product term), reuse LDS as f32 scratch
  __syncthreads();
  float* fsh = (float*)&sh[0][0][0];        // 16 groups x 256 cols
#pragma unroll
  for (int jj = 0; jj < 4; ++jj) fsh[g * DDIM + c4 + jj] = csum[jj];
  __syncthreads();
  if (t < DDIM) {
    float ssum = 0.f;
    for (int g2 = 0; g2 < 16; ++g2) ssum += fsh[g2 * DDIM + t];
    wsS[blockIdx.x * DDIM + t] = ssum;
  }
}

// ---------------------------------------------------------------------------
// Kernel 2: deterministic reduce of partials, M = G W + s b^T, store M
// transposed as hi/lo bf16 planes (mt[k][d]) for direct B-fragment staging.
// ---------------------------------------------------------------------------
__global__ __launch_bounds__(256, 1)
void reduce_m_kernel(const float* __restrict__ wsG, const float* __restrict__ wsS,
                     const float* __restrict__ Wm,  const float* __restrict__ bvec,
                     bf16* __restrict__ mtHi, bf16* __restrict__ mtLo) {
  __shared__ float grow[DDIM];
  __shared__ float sd;
  const int tid = threadIdx.x;
  const int d   = blockIdx.x;
  float g = 0.f;
  for (int p = 0; p < P_GRAM; ++p) g += wsG[(size_t)p * DDIM * DDIM + d * DDIM + tid];
  grow[tid] = g;
  if (tid == 0) {
    float s = 0.f;
    for (int p = 0; p < P_GRAM; ++p) s += wsS[p * DDIM + d];
    sd = s;
  }
  __syncthreads();
  float m = sd * bvec[tid];                 // s[d] * b[k]
  for (int e = 0; e < DDIM; ++e) m += grow[e] * Wm[e * DDIM + tid];
  bf16 hi = (bf16)m;
  mtHi[tid * DDIM + d] = hi;                // transposed: mt[k][d]
  mtLo[tid * DDIM + d] = (bf16)(m - (float)hi);
}

// ---------------------------------------------------------------------------
// Kernel 3: Out = Xf M.  grid = 128 blocks (256 rows each) x 1024 threads.
// M^T staged into LDS via CDNA5 async global->LDS DMA (ASYNCcnt), two
// 128-wide K halves; A fragments built on the fly from global fp32 x.
// ---------------------------------------------------------------------------
__global__ __launch_bounds__(1024, 1)
void out_kernel(const float* __restrict__ x,
                const bf16* __restrict__ mtHi, const bf16* __restrict__ mtLo,
                float* __restrict__ out) {
  __shared__ bf16 shm[2][DDIM][RS];
  const int t    = threadIdx.x;
  const int lane = t & 31;
  const int w    = t >> 5;
  const int mt   = w >> 1;
  const int tj0  = (w & 1) * 8;
  const int row0 = blockIdx.x * 256;
  const int m    = lane & 15, h = lane >> 4;

  v8f acc[8] = {};
  const float* arow = x + (size_t)(row0 + mt * 16 + m) * DDIM;

  // LDS byte addresses (low 32 bits of generic pointer == LDS offset)
  const uint ldsH = (uint)(uintptr_t)&shm[0][0][0];
  const uint ldsL = (uint)(uintptr_t)&shm[1][0][0];

  for (int kh = 0; kh < 2; ++kh) {
    __syncthreads();
    // async-stage mt[:, kh*128 .. +128) hi/lo planes (256B/row -> 272B rows)
#pragma unroll
    for (int i = 0; i < 4; ++i) {
      int idx  = t + 1024 * i;              // 0..4095 : 16B chunks
      int n    = idx >> 4;                  // matrix row 0..255
      int c16  = idx & 15;                  // chunk within row
      uint goff = (uint)(n * (DDIM * 2) + kh * 256 + c16 * 16);
      uint loff = (uint)(n * (RS * 2) + c16 * 16);
      uint dH = ldsH + loff, dL = ldsL + loff;
      asm volatile("global_load_async_to_lds_b128 %0, %1, %2 offset:0"
                   :: "v"(dH), "v"(goff), "s"(mtHi) : "memory");
      asm volatile("global_load_async_to_lds_b128 %0, %1, %2 offset:0"
                   :: "v"(dL), "v"(goff), "s"(mtLo) : "memory");
    }
    asm volatile("s_wait_asynccnt 0x0" ::: "memory");
    __syncthreads();
#pragma unroll
    for (int ks = 0; ks < 4; ++ks) {
      const float* p0 = arow + kh * 128 + ks * 32 + 8 * h;
      union { v16bf v; bf16 e[16]; } AH, AL;
      float4 fa = *(const float4*)(p0);
      float4 fb = *(const float4*)(p0 + 4);
      float4 fc = *(const float4*)(p0 + 16);
      float4 fd = *(const float4*)(p0 + 20);
      float fl[16] = {fa.x, fa.y, fa.z, fa.w, fb.x, fb.y, fb.z, fb.w,
                      fc.x, fc.y, fc.z, fc.w, fd.x, fd.y, fd.z, fd.w};
#pragma unroll
      for (int i = 0; i < 16; ++i) {
        float f = fl[i];
        bf16 hi = (bf16)f; AH.e[i] = hi; AL.e[i] = (bf16)(f - (float)hi);
      }
#pragma unroll
      for (int j = 0; j < 8; ++j) {
        v16bf bh = frag_b(&shm[0][(tj0 + j) * 16][ks * 32], lane);
        v16bf bl = frag_b(&shm[1][(tj0 + j) * 16][ks * 32], lane);
        acc[j] = wmma3(AH.v, AL.v, bh, bl, acc[j]);
      }
    }
  }

  int n = lane & 15;
#pragma unroll
  for (int j = 0; j < 8; ++j) {
    int col = (tj0 + j) * 16 + n;
#pragma unroll
    for (int q = 0; q < 8; ++q)
      out[(size_t)(row0 + mt * 16 + q + 8 * h) * DDIM + col] = acc[j][q];
  }
}

// ---------------------------------------------------------------------------
extern "C" void kernel_launch(void* const* d_in, const int* in_sizes, int n_in,
                              void* d_out, int out_size, void* d_ws, size_t ws_size,
                              hipStream_t stream) {
  const float* x  = (const float*)d_in[0];
  const float* Wm = (const float*)d_in[1];
  const float* b  = (const float*)d_in[2];
  float* out = (float*)d_out;

  // workspace layout (~17.1 MB): partial Grams | partial colsums | M^T hi/lo
  float* wsG = (float*)d_ws;
  float* wsS = wsG + (size_t)P_GRAM * DDIM * DDIM;
  bf16*  mtHi = (bf16*)(wsS + P_GRAM * DDIM);
  bf16*  mtLo = mtHi + DDIM * DDIM;

  hipLaunchKernelGGL(gram_kernel, dim3(P_GRAM), dim3(1024), 0, stream, x, wsG, wsS);
  hipLaunchKernelGGL(reduce_m_kernel, dim3(DDIM), dim3(DDIM), 0, stream,
                     wsG, wsS, Wm, b, mtHi, mtLo);
  hipLaunchKernelGGL(out_kernel, dim3(N_TOT / 256), dim3(1024), 0, stream,
                     x, mtHi, mtLo, out);
  (void)in_sizes; (void)n_in; (void)out_size; (void)ws_size;
}
